// FlashDeepseekV2Attention_39204461478528
// MI455X (gfx1250) — matrix operand order repaired and checked by reading, not compiled, not run on hardware
//
#include <hip/hip_runtime.h>
#include <hip/hip_bf16.h>

// ---------------------------------------------------------------------------
// Types for gfx1250 WMMA (wave32): A/B = 16 bf16 (8 VGPRs), C/D = 8 f32.
// ---------------------------------------------------------------------------
typedef __attribute__((ext_vector_type(16))) __bf16 v16bf;
typedef __attribute__((ext_vector_type(8)))  float  v8f;
typedef __attribute__((ext_vector_type(8)))  __bf16 bf16x8;
typedef __attribute__((ext_vector_type(4)))  float  f32x4;
typedef __attribute__((ext_vector_type(4)))  unsigned int u32x4;
typedef __attribute__((ext_vector_type(8)))  int  i32x8;
typedef __attribute__((ext_vector_type(4)))  int  i32x4;

// Tensor Data Mover availability (device pass only).
#if defined(__HIP_DEVICE_COMPILE__) && defined(__has_builtin)
#if __has_builtin(__builtin_amdgcn_tensor_load_to_lds) && __has_builtin(__builtin_amdgcn_s_wait_tensorcnt)
#define HAVE_TDM 1
#endif
#endif
#ifndef HAVE_TDM
#define HAVE_TDM 0
#endif

__device__ __forceinline__ v8f wmma_bf16(v16bf a, v16bf b, v8f c) {
    return __builtin_amdgcn_wmma_f32_16x16x32_bf16(
        false, a, false, b, (short)0, c, false, false);
}

__device__ __forceinline__ v16bf pack16(bf16x8 lo, bf16x8 hi) {
    v16bf r;
#pragma unroll
    for (int i = 0; i < 8; ++i) { r[i] = lo[i]; r[i + 8] = hi[i]; }
    return r;
}

#if HAVE_TDM
// ---------------------------------------------------------------------------
// TDM 2D tile load: bf16 tile of [rows x 32] elements, row stride
// `strideElems`, into LDS at byte offset `ldsOff`, inserting a 16B pad after
// every 64B row (matches the G_LDK=40-half padded LDS rows).
// D# layout per CDNA5 ISA ch.8 (group0 128b, group1 256b; groups 2/3 zero).
// Caller guards to a single wave via a *scalar* branch; EXEC is ignored.
// ---------------------------------------------------------------------------
__device__ __forceinline__ void tdm_load_tile(unsigned ldsOff, const __bf16* g,
                                              int rows, long strideElems)
{
    const unsigned long long ga = (unsigned long long)(size_t)g;
    u32x4 g0;
    g0[0] = 1u;                                    // count=1, user descriptor
    g0[1] = ldsOff;                                // lds_addr (bytes)
    g0[2] = (unsigned)(ga & 0xffffffffull);        // global_addr[31:0]
    g0[3] = (unsigned)((ga >> 32) & 0x1ffffffull)  // global_addr[56:32]
          | (2u << 30);                            // type=2 ("image")
    const unsigned td0 = (unsigned)strideElems;    // tensor_dim0 (OOB bound)
    const unsigned td1 = 1u << 20;                 // tensor_dim1 (OOB bound)
    const unsigned long long st = (unsigned long long)strideElems;
    i32x8 g1;
    g1[0] = (1 << 16)        // data_size = 1 -> 2 bytes
          | (1 << 20)        // pad_enable
          | (3 << 22)        // pad_interval code 3 -> every 16 DWORDs (64B)
          | (3 << 25);       // pad_amount  code 3 -> 4 DWORDs (16B)
    g1[1] = (int)((td0 & 0xffffu) << 16);                        // dim0[15:0]
    g1[2] = (int)((td0 >> 16) & 0xffffu) | (int)((td1 & 0xffffu) << 16);
    g1[3] = (int)((td1 >> 16) & 0xffffu) | (32 << 16);           // tile_dim0=32
    g1[4] = rows & 0xffff;                                       // tile_dim1
    g1[5] = (int)(st & 0xffffffffull);                           // dim0_stride lo
    g1[6] = (int)((st >> 32) & 0xffffu);                         // stride hi
    g1[7] = 0;
    i32x4 z; z[0] = 0; z[1] = 0; z[2] = 0; z[3] = 0;
#if __clang_major__ >= 23
    i32x8 z8;
#pragma unroll
    for (int i = 0; i < 8; ++i) z8[i] = 0;
    __builtin_amdgcn_tensor_load_to_lds(g0, g1, z, z, z8, 0);
#else
    __builtin_amdgcn_tensor_load_to_lds(g0, g1, z, z, 0);
#endif
}
#endif  // HAVE_TDM

// ---------------------------------------------------------------------------
// GEMM: C[M,N] (f32) = A[M,K] (bf16, row-major) x BT[N,K] (bf16, B transposed)
// WG = 128 threads (4 waves), WG tile 64x64, wave tile 32x32, K-step 32.
// Double-buffered LDS: step s computes on buf while the *next* tile streams
// into 1-buf (TDM issued by wave 0 through a scalar branch; fallback stages
// through registers). One barrier per K-step.
// ---------------------------------------------------------------------------
#define G_LDK 40   // padded K stride in halfs (32 + 8) => 80B rows, 16B aligned

__global__ __launch_bounds__(128)
void gemm_bf16wmma(const __bf16* __restrict__ A, const __bf16* __restrict__ BT,
                   float* __restrict__ C, int M, int N, int K)
{
    __shared__ __bf16 As[2][64][G_LDK];   // [buf][m][k]
    __shared__ __bf16 Bs[2][64][G_LDK];   // [buf][n][k]

    const int t    = threadIdx.x;
    const int lane = t & 31;
    const int wave = t >> 5;
    const int lh   = lane & 15;
    const bool hih = lane >= 16;

    const int m0 = blockIdx.y * 64;
    const int n0 = blockIdx.x * 64;
    const int wm = (wave >> 1) * 32;
    const int wn = (wave & 1) * 32;

    v8f acc[2][2];
#pragma unroll
    for (int i = 0; i < 2; ++i)
#pragma unroll
        for (int j = 0; j < 2; ++j)
#pragma unroll
            for (int r = 0; r < 8; ++r) acc[i][j][r] = 0.0f;

    const int cpRow = t >> 1;          // 0..63
    const int cpOff = (t & 1) * 16;    // 0 / 16 halfs

    // stage step 0 into buffer 0 (compiler-visible LDS stores)
    {
        const __bf16* asrc = A  + (size_t)(m0 + cpRow) * K + cpOff;
        const __bf16* bsrc = BT + (size_t)(n0 + cpRow) * K + cpOff;
        *(bf16x8*)&As[0][cpRow][cpOff]     = *(const bf16x8*)(asrc);
        *(bf16x8*)&As[0][cpRow][cpOff + 8] = *(const bf16x8*)(asrc + 8);
        *(bf16x8*)&Bs[0][cpRow][cpOff]     = *(const bf16x8*)(bsrc);
        *(bf16x8*)&Bs[0][cpRow][cpOff + 8] = *(const bf16x8*)(bsrc + 8);
    }
    __syncthreads();

    const int nsteps = K >> 5;
    const int aOff = hih ? 8 : 0;    // A frag: K = {aOff..+7, aOff+16..+23}
    const int bOff = hih ? 8 : 0;    // B frag: K = {bOff..+7, bOff+16..+23}

#if HAVE_TDM
    // force the wave id into an SGPR so the TDM guard is a scalar branch
    // (single DMA issue per workgroup; TDM ignores EXEC)
    const int uwave = __builtin_amdgcn_readfirstlane(wave);
#endif

    for (int s = 0; s < nsteps; ++s) {
        const int buf = s & 1;
        // ---- prefetch next K-tile into the other buffer --------------------
        if (s + 1 < nsteps) {
            const int kb = (s + 1) * 32;
#if HAVE_TDM
            if (uwave == 0) {
                tdm_load_tile((unsigned)(size_t)&As[1 - buf][0][0],
                              A + (size_t)m0 * K + kb, 64, (long)K);
                tdm_load_tile((unsigned)(size_t)&Bs[1 - buf][0][0],
                              BT + (size_t)n0 * K + kb, 64, (long)K);
            }
#else
            const __bf16* asrc = A  + (size_t)(m0 + cpRow) * K + kb + cpOff;
            const __bf16* bsrc = BT + (size_t)(n0 + cpRow) * K + kb + cpOff;
            *(bf16x8*)&As[1 - buf][cpRow][cpOff]     = *(const bf16x8*)(asrc);
            *(bf16x8*)&As[1 - buf][cpRow][cpOff + 8] = *(const bf16x8*)(asrc + 8);
            *(bf16x8*)&Bs[1 - buf][cpRow][cpOff]     = *(const bf16x8*)(bsrc);
            *(bf16x8*)&Bs[1 - buf][cpRow][cpOff + 8] = *(const bf16x8*)(bsrc + 8);
#endif
        }

        // ---- compute on current buffer -------------------------------------
        v16bf afr[2], bfr[2];
#pragma unroll
        for (int i = 0; i < 2; ++i) {
            const int row = wm + 16 * i + lh;
            afr[i] = pack16(*(const bf16x8*)&As[buf][row][aOff],
                            *(const bf16x8*)&As[buf][row][aOff + 16]);
        }
#pragma unroll
        for (int j = 0; j < 2; ++j) {
            const int col = wn + 16 * j + lh;
            bfr[j] = pack16(*(const bf16x8*)&Bs[buf][col][bOff],
                            *(const bf16x8*)&Bs[buf][col][bOff + 16]);
        }
#pragma unroll
        for (int i = 0; i < 2; ++i)
#pragma unroll
            for (int j = 0; j < 2; ++j)
                acc[i][j] = wmma_bf16(afr[i], bfr[j], acc[i][j]);

#if HAVE_TDM
        if (uwave == 0) __builtin_amdgcn_s_wait_tensorcnt(0);
#endif
        __syncthreads();   // next buffer ready for everyone; current released
    }

    const int rofs = hih ? 8 : 0;
#pragma unroll
    for (int i = 0; i < 2; ++i)
#pragma unroll
        for (int j = 0; j < 2; ++j) {
            float* cp = C + (size_t)(m0 + wm + 16 * i + rofs) * N + (n0 + wn + 16 * j + lh);
#pragma unroll
            for (int r = 0; r < 8; ++r)
                cp[(size_t)r * N] = acc[i][j][r];
        }
}

// ---------------------------------------------------------------------------
// Weight convert + transpose: W[K][N] f32 -> WT[N][K] bf16 (32x32 LDS tiles).
// ---------------------------------------------------------------------------
__global__ void convert_transpose(const float* __restrict__ W, __bf16* __restrict__ WT,
                                  int K, int N)
{
    __shared__ float tile[32][33];
    const int k0 = blockIdx.y * 32, n0 = blockIdx.x * 32;
    const int tx = threadIdx.x, ty = threadIdx.y;   // (32, 8)
#pragma unroll
    for (int r = 0; r < 32; r += 8)
        tile[ty + r][tx] = W[(size_t)(k0 + ty + r) * N + n0 + tx];
    __syncthreads();
#pragma unroll
    for (int r = 0; r < 32; r += 8)
        WT[(size_t)(n0 + ty + r) * K + k0 + tx] = (__bf16)tile[tx][ty + r];
}

__global__ void convert_bf16(const float* __restrict__ X, __bf16* __restrict__ Y, size_t n)
{
    const size_t i = (size_t)blockIdx.x * blockDim.x + threadIdx.x;
    if (i < n) Y[i] = (__bf16)X[i];
}

// ---------------------------------------------------------------------------
// RMSNorm rows (f32 in, bf16 out). One 256-thread block per row.
// ---------------------------------------------------------------------------
__global__ void rmsnorm_rows(const float* __restrict__ X, const float* __restrict__ w,
                             __bf16* __restrict__ Y, int W)
{
    const int row = blockIdx.x;
    const float* x = X + (size_t)row * W;
    float ss = 0.0f;
    for (int j = threadIdx.x; j < W; j += blockDim.x) { float v = x[j]; ss += v * v; }
    __shared__ float red[256];
    red[threadIdx.x] = ss;
    __syncthreads();
    for (int s = 128; s > 0; s >>= 1) {
        if (threadIdx.x < s) red[threadIdx.x] += red[threadIdx.x + s];
        __syncthreads();
    }
    const float inv = rsqrtf(red[0] / (float)W + 1e-6f);
    for (int j = threadIdx.x; j < W; j += blockDim.x)
        Y[(size_t)row * W + j] = (__bf16)(x[j] * inv * w[j]);
}

__device__ __forceinline__ float rope_angle(int s, int i /*0..31*/) {
    const float invf = exp2f(-(float)i * (13.287712379549449f / 32.0f));
    return (float)s * invf;
}

// ---------------------------------------------------------------------------
// ckv[2048,576] f32 -> kvcn bf16 [2048,512] (rmsnorm), kpe bf16 [2048,64] (RoPE)
// ---------------------------------------------------------------------------
__global__ void kv_prep(const float* __restrict__ ckv, const float* __restrict__ w,
                        __bf16* __restrict__ kvcn, __bf16* __restrict__ kpe)
{
    const int s = blockIdx.x;
    const float* row = ckv + (size_t)s * 576;
    float ss = 0.0f;
    for (int j = threadIdx.x; j < 512; j += 128) { float v = row[j]; ss += v * v; }
    __shared__ float red[128];
    red[threadIdx.x] = ss;
    __syncthreads();
    for (int t = 64; t > 0; t >>= 1) {
        if (threadIdx.x < t) red[threadIdx.x] += red[threadIdx.x + t];
        __syncthreads();
    }
    const float inv = rsqrtf(red[0] / 512.0f + 1e-6f);
    for (int j = threadIdx.x; j < 512; j += 128)
        kvcn[(size_t)s * 512 + j] = (__bf16)(row[j] * inv * w[j]);
    if (threadIdx.x < 64) {
        const int j = threadIdx.x;
        const float ang = rope_angle(s, j & 31);
        const float c = cosf(ang), sn = sinf(ang);
        const float x = row[512 + j];
        const float rot = (j < 32) ? -row[512 + j + 32] : row[512 + j - 32];
        kpe[(size_t)s * 64 + j] = (__bf16)(x * c + rot * sn);
    }
}

// ---------------------------------------------------------------------------
// q_raw [s][h*192] f32 -> Qb bf16 [h][s][192] with RoPE on last 64 dims
// ---------------------------------------------------------------------------
__global__ void q_prep(const float* __restrict__ qraw, __bf16* __restrict__ Qb)
{
    const int s = blockIdx.x, h = blockIdx.y, j = threadIdx.x;   // j in [0,192)
    const float* src = qraw + ((size_t)s * 32 + h) * 192;
    float v;
    if (j < 128) {
        v = src[j];
    } else {
        const int jp = j - 128;
        const float ang = rope_angle(s, jp & 31);
        const float c = cosf(ang), sn = sinf(ang);
        const float x = src[128 + jp];
        const float rot = (jp < 32) ? -src[128 + jp + 32] : src[128 + jp - 32];
        v = x * c + rot * sn;
    }
    Qb[((size_t)h * 2048 + s) * 192 + j] = (__bf16)v;
}

// ---------------------------------------------------------------------------
// kv_raw [s][h*256] f32 + kpe -> Kb bf16 [h][s][192], Vb bf16 [h][s][128]
// ---------------------------------------------------------------------------
__global__ void kv_pack(const float* __restrict__ kvraw, const __bf16* __restrict__ kpe,
                        __bf16* __restrict__ Kb, __bf16* __restrict__ Vb)
{
    const int s = blockIdx.x, h = blockIdx.y, j = threadIdx.x;   // j in [0,192)
    const float* src = kvraw + ((size_t)s * 32 + h) * 256;
    const size_t kbase = ((size_t)h * 2048 + s) * 192;
    if (j < 128) {
        Kb[kbase + j] = (__bf16)src[j];
        Vb[((size_t)h * 2048 + s) * 128 + j] = (__bf16)src[128 + j];
    } else {
        Kb[kbase + j] = kpe[(size_t)s * 64 + (j - 128)];
    }
}

// ---------------------------------------------------------------------------
// Causal flash attention. Grid (qTiles=32, heads=32), 128 threads (4 waves).
// Wave w: queries q0+16w..+15. K/V tiles of 32 keys staged in LDS.
// Scores via 6 chained WMMAs over d=192; P re-laid via per-wave LDS bounce.
// Output written directly as bf16 for the final WMMA GEMM.
// ---------------------------------------------------------------------------
#define A_LDK 200
#define A_LDV 40
#define A_LDP 40

__global__ __launch_bounds__(128)
void mla_attention(const __bf16* __restrict__ Qb, const __bf16* __restrict__ Kb,
                   const __bf16* __restrict__ Vb, __bf16* __restrict__ O)
{
    __shared__ __bf16 Kt[32][A_LDK];
    __shared__ __bf16 Vt[128][A_LDV];
    __shared__ __bf16 Pt[4][16][A_LDP];

    const int t    = threadIdx.x;
    const int lane = t & 31;
    const int wave = t >> 5;
    const int lh   = lane & 15;
    const bool hih = lane >= 16;
    const int h  = blockIdx.y;
    const int q0 = blockIdx.x * 64;
    const int qw = q0 + wave * 16;

    v16bf qf[6];
    {
        const __bf16* qrow = Qb + ((size_t)h * 2048 + (qw + lh)) * 192;
#pragma unroll
        for (int c = 0; c < 6; ++c) {
            const int kOff = c * 32 + (hih ? 8 : 0);
            qf[c] = pack16(*(const bf16x8*)(qrow + kOff),
                           *(const bf16x8*)(qrow + kOff + 16));
        }
    }

    v8f acc[8];
#pragma unroll
    for (int j = 0; j < 8; ++j)
#pragma unroll
        for (int r = 0; r < 8; ++r) acc[j][r] = 0.0f;
    float mrun[8], lrun[8];
#pragma unroll
    for (int r = 0; r < 8; ++r) { mrun[r] = -1e30f; lrun[r] = 0.0f; }

    const float scale = 0.07216878364870323f;   // 192^-0.5
    const float L2E   = 1.4426950408889634f;

    const int ldKey = t >> 2;
    const int ldSeg = t & 3;

    for (int t0 = 0; t0 <= q0 + 63; t0 += 32) {
        {
            const __bf16* ksrc = Kb + ((size_t)h * 2048 + (t0 + ldKey)) * 192 + ldSeg * 48;
#pragma unroll
            for (int u = 0; u < 6; ++u)
                *(bf16x8*)&Kt[ldKey][ldSeg * 48 + u * 8] = *(const bf16x8*)(ksrc + u * 8);
        }
        {
            const __bf16* vsrc = Vb + ((size_t)h * 2048 + (t0 + ldKey)) * 128 + ldSeg * 32;
            bf16x8 vv[4];
#pragma unroll
            for (int u = 0; u < 4; ++u) vv[u] = *(const bf16x8*)(vsrc + u * 8);
#pragma unroll
            for (int u = 0; u < 4; ++u)
#pragma unroll
                for (int e = 0; e < 8; ++e)
                    Vt[ldSeg * 32 + u * 8 + e][ldKey] = vv[u][e];
        }
        // prefetch next tiles into caches (global_prefetch_b8)
        if (t0 + 32 <= q0 + 63) {
            __builtin_prefetch(Kb + ((size_t)h * 2048 + (t0 + 32 + ldKey)) * 192 + ldSeg * 48, 0, 1);
            __builtin_prefetch(Vb + ((size_t)h * 2048 + (t0 + 32 + ldKey)) * 128 + ldSeg * 32, 0, 1);
        }
        __syncthreads();

        v8f s0, s1;
#pragma unroll
        for (int r = 0; r < 8; ++r) { s0[r] = 0.0f; s1[r] = 0.0f; }
#pragma unroll
        for (int c = 0; c < 6; ++c) {
            const int dOff = c * 32 + (hih ? 8 : 0);
            v16bf b0 = pack16(*(const bf16x8*)&Kt[lh][dOff],
                              *(const bf16x8*)&Kt[lh][dOff + 16]);
            v16bf b1 = pack16(*(const bf16x8*)&Kt[16 + lh][dOff],
                              *(const bf16x8*)&Kt[16 + lh][dOff + 16]);
            s0 = wmma_bf16(qf[c], b0, s0);
            s1 = wmma_bf16(qf[c], b1, s1);
        }

        float sm0[8], sm1[8];
        const int qbase = qw + (hih ? 8 : 0);
#pragma unroll
        for (int r = 0; r < 8; ++r) {
            const int qq = qbase + r;
            sm0[r] = (t0 + lh      <= qq) ? s0[r] * scale : -1e30f;
            sm1[r] = (t0 + 16 + lh <= qq) ? s1[r] * scale : -1e30f;
        }

        float alpha[8], mnew[8];
#pragma unroll
        for (int r = 0; r < 8; ++r) {
            float mx = fmaxf(sm0[r], sm1[r]);
#pragma unroll
            for (int m = 1; m < 16; m <<= 1) mx = fmaxf(mx, __shfl_xor(mx, m, 32));
            mnew[r]  = fmaxf(mrun[r], mx);
            alpha[r] = exp2f((mrun[r] - mnew[r]) * L2E);
            mrun[r]  = mnew[r];
        }
#pragma unroll
        for (int r = 0; r < 8; ++r) {
            const float e0 = exp2f((sm0[r] - mnew[r]) * L2E);
            const float e1 = exp2f((sm1[r] - mnew[r]) * L2E);
            const int qr = r + (hih ? 8 : 0);
            Pt[wave][qr][lh]      = (__bf16)e0;
            Pt[wave][qr][16 + lh] = (__bf16)e1;
            float sum = e0 + e1;
#pragma unroll
            for (int m = 1; m < 16; m <<= 1) sum += __shfl_xor(sum, m, 32);
            lrun[r] = lrun[r] * alpha[r] + sum;
        }

        v16bf pa;
        {
            const int tOff = hih ? 8 : 0;
            pa = pack16(*(const bf16x8*)&Pt[wave][lh][tOff],
                        *(const bf16x8*)&Pt[wave][lh][tOff + 16]);
        }
        const int vOff = hih ? 8 : 0;
#pragma unroll
        for (int j = 0; j < 8; ++j) {
#pragma unroll
            for (int r = 0; r < 8; ++r) acc[j][r] *= alpha[r];
            const int dv = j * 16 + lh;
            v16bf vb = pack16(*(const bf16x8*)&Vt[dv][vOff],
                              *(const bf16x8*)&Vt[dv][vOff + 16]);
            acc[j] = wmma_bf16(pa, vb, acc[j]);
        }
        __syncthreads();
    }

#pragma unroll
    for (int j = 0; j < 8; ++j)
#pragma unroll
        for (int r = 0; r < 8; ++r) {
            const int q  = qw + r + (hih ? 8 : 0);
            const int dv = j * 16 + lh;
            O[(size_t)q * 4096 + h * 128 + dv] = (__bf16)(acc[j][r] / lrun[r]);
        }
}

// ---------------------------------------------------------------------------
// Host-side launch
// ---------------------------------------------------------------------------
extern "C" void kernel_launch(void* const* d_in, const int* in_sizes, int n_in,
                              void* d_out, int out_size, void* d_ws, size_t ws_size,
                              hipStream_t stream)
{
    const float* hidden = (const float*)d_in[0];   // [1,2048,5120]
    const float* w_qa   = (const float*)d_in[1];   // [5120,1536]
    const float* qa_ln  = (const float*)d_in[2];   // [1536]
    const float* w_qb   = (const float*)d_in[3];   // [1536,6144]
    const float* w_kva  = (const float*)d_in[4];   // [5120,576]
    const float* kva_ln = (const float*)d_in[5];   // [512]
    const float* w_kvb  = (const float*)d_in[6];   // [512,8192]
    const float* w_o    = (const float*)d_in[7];   // [4096,5120]
    float* out = (float*)d_out;                    // [2048,5120]

    char* ws = (char*)d_ws;
    size_t off = 0;
    auto alloc = [&](size_t bytes) -> void* {
        void* p = ws + off;
        off += (bytes + 255) & ~(size_t)255;
        return p;
    };

    // bf16 transposed weights (converted once; reused 32x each by GEMMs)
    __bf16* wqaT  = (__bf16*)alloc(1536ull * 5120 * 2);
    __bf16* wqbT  = (__bf16*)alloc(6144ull * 1536 * 2);
    __bf16* wkvaT = (__bf16*)alloc(576ull  * 5120 * 2);
    __bf16* wkvbT = (__bf16*)alloc(8192ull * 512  * 2);
    __bf16* woT   = (__bf16*)alloc(5120ull * 4096 * 2);
    __bf16* hidb  = (__bf16*)alloc(2048ull * 5120 * 2);
    float*  qa_raw = (float*) alloc(2048ull * 1536 * 4);
    __bf16* qa_n   = (__bf16*)alloc(2048ull * 1536 * 2);
    float*  ckv    = (float*) alloc(2048ull * 576  * 4);
    __bf16* kvc_n  = (__bf16*)alloc(2048ull * 512  * 2);
    __bf16* kpe    = (__bf16*)alloc(2048ull * 64   * 2);
    float*  q_raw  = (float*) alloc(2048ull * 6144 * 4);
    float*  kv_raw = (float*) alloc(2048ull * 8192 * 4);
    __bf16* Qb     = (__bf16*)alloc(32ull * 2048 * 192 * 2);
    __bf16* Kb     = (__bf16*)alloc(32ull * 2048 * 192 * 2);
    __bf16* Vb     = (__bf16*)alloc(32ull * 2048 * 128 * 2);
    __bf16* attnb  = (__bf16*)alloc(2048ull * 4096 * 2);

    const dim3 tblk(32, 8);
    convert_transpose<<<dim3(1536 / 32, 5120 / 32), tblk, 0, stream>>>(w_qa,  wqaT,  5120, 1536);
    convert_transpose<<<dim3(6144 / 32, 1536 / 32), tblk, 0, stream>>>(w_qb,  wqbT,  1536, 6144);
    convert_transpose<<<dim3(576  / 32, 5120 / 32), tblk, 0, stream>>>(w_kva, wkvaT, 5120, 576);
    convert_transpose<<<dim3(8192 / 32, 512  / 32), tblk, 0, stream>>>(w_kvb, wkvbT, 512,  8192);
    convert_transpose<<<dim3(5120 / 32, 4096 / 32), tblk, 0, stream>>>(w_o,   woT,   4096, 5120);
    {
        const size_t n = 2048ull * 5120;
        convert_bf16<<<(unsigned)((n + 255) / 256), 256, 0, stream>>>(hidden, hidb, n);
    }

    const dim3 gblk(128);

    gemm_bf16wmma<<<dim3(1536 / 64, 2048 / 64), gblk, 0, stream>>>(hidb, wqaT, qa_raw, 2048, 1536, 5120);
    rmsnorm_rows<<<2048, 256, 0, stream>>>(qa_raw, qa_ln, qa_n, 1536);

    gemm_bf16wmma<<<dim3(576 / 64, 2048 / 64), gblk, 0, stream>>>(hidb, wkvaT, ckv, 2048, 576, 5120);
    kv_prep<<<2048, 128, 0, stream>>>(ckv, kva_ln, kvc_n, kpe);

    gemm_bf16wmma<<<dim3(6144 / 64, 2048 / 64), gblk, 0, stream>>>(qa_n, wqbT, q_raw, 2048, 6144, 1536);
    q_prep<<<dim3(2048, 32), 192, 0, stream>>>(q_raw, Qb);

    gemm_bf16wmma<<<dim3(8192 / 64, 2048 / 64), gblk, 0, stream>>>(kvc_n, wkvbT, kv_raw, 2048, 8192, 512);
    kv_pack<<<dim3(2048, 32), 192, 0, stream>>>(kv_raw, kpe, Kb, Vb);

    mla_attention<<<dim3(32, 32), 128, 0, stream>>>(Qb, Kb, Vb, attnb);

    gemm_bf16wmma<<<dim3(5120 / 64, 2048 / 64), gblk, 0, stream>>>(attnb, woT, out, 2048, 5120, 4096);
}
